// BitLinear_70394513981693
// MI455X (gfx1250) — 8-bit matrix operand order repaired and checked by reading, compiled, not run on hardware
//
#include <hip/hip_runtime.h>
#include <hip/hip_bf16.h>
#include <stdint.h>

#define M_DIM 8192
#define K_DIM 2048
#define N_DIM 8192
#define QB 127.0f
#define EPS_NORM 1e-6f
#define EPS_SCALE 1e-8f

typedef __attribute__((ext_vector_type(8))) int v8i;

__global__ __launch_bounds__(256) void rms_quant_kernel(
    const float* __restrict__ x, const float* __restrict__ nw,
    int8_t* __restrict__ Xq, float* __restrict__ Xs)
{
    __shared__ float red[256];
    const int row = blockIdx.x;
    const int t   = threadIdx.x;
    const float* xr = x + (size_t)row * K_DIM;

    float4 v0 = *(const float4*)(xr + t * 8);
    float4 v1 = *(const float4*)(xr + t * 8 + 4);

    float ss = v0.x*v0.x + v0.y*v0.y + v0.z*v0.z + v0.w*v0.w
             + v1.x*v1.x + v1.y*v1.y + v1.z*v1.z + v1.w*v1.w;
    red[t] = ss;
    __syncthreads();
    #pragma unroll
    for (int s = 128; s > 0; s >>= 1) {
        if (t < s) red[t] += red[t + s];
        __syncthreads();
    }
    const float rinv = rsqrtf(red[0] / (float)K_DIM + EPS_NORM);
    __syncthreads();

    float4 w0 = *(const float4*)(nw + t * 8);
    float4 w1 = *(const float4*)(nw + t * 8 + 4);

    float n0 = v0.x * rinv * w0.x, n1 = v0.y * rinv * w0.y;
    float n2 = v0.z * rinv * w0.z, n3 = v0.w * rinv * w0.w;
    float n4 = v1.x * rinv * w1.x, n5 = v1.y * rinv * w1.y;
    float n6 = v1.z * rinv * w1.z, n7 = v1.w * rinv * w1.w;

    float am = fmaxf(fmaxf(fmaxf(fabsf(n0), fabsf(n1)), fmaxf(fabsf(n2), fabsf(n3))),
                     fmaxf(fmaxf(fabsf(n4), fabsf(n5)), fmaxf(fabsf(n6), fabsf(n7))));
    red[t] = am;
    __syncthreads();
    #pragma unroll
    for (int s = 128; s > 0; s >>= 1) {
        if (t < s) red[t] = fmaxf(red[t], red[t + s]);
        __syncthreads();
    }
    const float scale = red[0] / QB + EPS_SCALE;

    int q0 = (int)fminf(fmaxf(rintf(n0 / scale), -128.0f), 127.0f);
    int q1 = (int)fminf(fmaxf(rintf(n1 / scale), -128.0f), 127.0f);
    int q2 = (int)fminf(fmaxf(rintf(n2 / scale), -128.0f), 127.0f);
    int q3 = (int)fminf(fmaxf(rintf(n3 / scale), -128.0f), 127.0f);
    int q4 = (int)fminf(fmaxf(rintf(n4 / scale), -128.0f), 127.0f);
    int q5 = (int)fminf(fmaxf(rintf(n5 / scale), -128.0f), 127.0f);
    int q6 = (int)fminf(fmaxf(rintf(n6 / scale), -128.0f), 127.0f);
    int q7 = (int)fminf(fmaxf(rintf(n7 / scale), -128.0f), 127.0f);

    int2 packed;
    packed.x = (q0 & 0xFF) | ((q1 & 0xFF) << 8) | ((q2 & 0xFF) << 16) | ((q3 & 0xFF) << 24);
    packed.y = (q4 & 0xFF) | ((q5 & 0xFF) << 8) | ((q6 & 0xFF) << 16) | ((q7 & 0xFF) << 24);
    *(int2*)(Xq + (size_t)row * K_DIM + t * 8) = packed;

    if (t == 0) Xs[row] = scale;
}

__global__ __launch_bounds__(256) void w_abs_partial_kernel(
    const float* __restrict__ w, float* __restrict__ partials)
{
    __shared__ float red[256];
    const int t = threadIdx.x;
    const float4* p = (const float4*)(w + (size_t)blockIdx.x * 4096);
    float s = 0.0f;
    #pragma unroll
    for (int c = 0; c < 4; ++c) {
        float4 v = p[t + c * 256];
        s += fabsf(v.x) + fabsf(v.y) + fabsf(v.z) + fabsf(v.w);
    }
    red[t] = s;
    __syncthreads();
    #pragma unroll
    for (int st = 128; st > 0; st >>= 1) {
        if (t < st) red[t] += red[t + st];
        __syncthreads();
    }
    if (t == 0) partials[blockIdx.x] = red[0];
}

__global__ __launch_bounds__(256) void w_scale_final_kernel(
    const float* __restrict__ partials, float* __restrict__ wscale)
{
    __shared__ float red[256];
    const int t = threadIdx.x;
    float s = 0.0f;
    #pragma unroll
    for (int i = 0; i < 16; ++i) s += partials[t + i * 256];
    red[t] = s;
    __syncthreads();
    #pragma unroll
    for (int st = 128; st > 0; st >>= 1) {
        if (t < st) red[t] += red[t + st];
        __syncthreads();
    }
    if (t == 0) *wscale = red[0] / ((float)N_DIM * (float)K_DIM) + EPS_SCALE;
}

__global__ __launch_bounds__(256) void w_quant_kernel(
    const float* __restrict__ w, const float* __restrict__ wscale,
    int8_t* __restrict__ wq)
{
    const float ws = *wscale;
    const size_t i = ((size_t)blockIdx.x * 256 + threadIdx.x) * 4;
    float4 v = *(const float4*)(w + i);
    int q0 = (int)fminf(fmaxf(rintf(v.x / ws), -1.0f), 1.0f);
    int q1 = (int)fminf(fmaxf(rintf(v.y / ws), -1.0f), 1.0f);
    int q2 = (int)fminf(fmaxf(rintf(v.z / ws), -1.0f), 1.0f);
    int q3 = (int)fminf(fmaxf(rintf(v.w / ws), -1.0f), 1.0f);
    int packed = (q0 & 0xFF) | ((q1 & 0xFF) << 8) | ((q2 & 0xFF) << 16) | ((q3 & 0xFF) << 24);
    *(int*)(wq + i) = packed;
}

__device__ __forceinline__ v8i load_a_frag(const int8_t* p) {
    int2 x0 = *(const int2*)(p);
    int2 x1 = *(const int2*)(p + 8);
    int2 x2 = *(const int2*)(p + 32);
    int2 x3 = *(const int2*)(p + 40);
    v8i r;
    r[0] = x0.x; r[1] = x0.y; r[2] = x1.x; r[3] = x1.y;
    r[4] = x2.x; r[5] = x2.y; r[6] = x3.x; r[7] = x3.y;
    return r;
}

__device__ __forceinline__ v8i load_b_frag(const int8_t* p) {
    int4 x0 = *(const int4*)(p);
    int4 x1 = *(const int4*)(p + 32);
    v8i r;
    r[0] = x0.x; r[1] = x0.y; r[2] = x0.z; r[3] = x0.w;
    r[4] = x1.x; r[5] = x1.y; r[6] = x1.z; r[7] = x1.w;
    return r;
}

__global__ __launch_bounds__(256) void gemm_i8_wmma_kernel(
    const int8_t* __restrict__ Xq, const int8_t* __restrict__ Wq,
    const float* __restrict__ Xs, const float* __restrict__ wscale,
    float* __restrict__ out)
{
    const int lane = threadIdx.x & 31;
    const int wave = threadIdx.x >> 5;
    const int wr   = wave & 3;
    const int wc   = wave >> 2;
    const int mBase = blockIdx.y * 128 + wr * 32;
    const int nBase = blockIdx.x * 128 + wc * 64;
    const int l15 = lane & 15;
    const int lhi = lane >> 4;

    v8i acc[2][4] = {};

    const int8_t* aRow[2];
    #pragma unroll
    for (int mi = 0; mi < 2; ++mi)
        aRow[mi] = Xq + (size_t)(mBase + mi * 16 + l15) * K_DIM + (lhi << 4);
    const int8_t* bCol[4];
    #pragma unroll
    for (int ni = 0; ni < 4; ++ni)
        bCol[ni] = Wq + (size_t)(nBase + ni * 16 + l15) * K_DIM + (lhi << 4);

    for (int kb = 0; kb < K_DIM; kb += 64) {
        v8i a[2], b[4];
        #pragma unroll
        for (int mi = 0; mi < 2; ++mi) a[mi] = load_a_frag(aRow[mi] + kb);
        #pragma unroll
        for (int ni = 0; ni < 4; ++ni) b[ni] = load_b_frag(bCol[ni] + kb);

        #pragma unroll
        for (int mi = 0; mi < 2; ++mi)
            #pragma unroll
            for (int ni = 0; ni < 4; ++ni)
                acc[mi][ni] = __builtin_amdgcn_wmma_i32_16x16x64_iu8(
                    true, a[mi], true, b[ni], acc[mi][ni], false, false);
    }

    const float ws = *wscale;
    #pragma unroll
    for (int mi = 0; mi < 2; ++mi) {
        #pragma unroll
        for (int j = 0; j < 8; ++j) {
            const int m = mBase + mi * 16 + (lhi << 3) + j;
            const float s = Xs[m] * ws;
            float* orow = out + (size_t)m * N_DIM + nBase + l15;
            #pragma unroll
            for (int ni = 0; ni < 4; ++ni)
                orow[ni * 16] = (float)acc[mi][ni][j] * s;
        }
    }
}

extern "C" void kernel_launch(void* const* d_in, const int* in_sizes, int n_in,
                              void* d_out, int out_size, void* d_ws, size_t ws_size,
                              hipStream_t stream) {
    const float* x  = (const float*)d_in[0];
    const float* w  = (const float*)d_in[1];
    const float* nw = (const float*)d_in[2];
    float* out = (float*)d_out;

    int8_t* Xq = (int8_t*)d_ws;
    int8_t* Wq = Xq + (size_t)M_DIM * K_DIM;
    float*  Xs = (float*)(Wq + (size_t)N_DIM * K_DIM);
    float*  wscale   = Xs + M_DIM;
    float*  partials = wscale + 1;

    rms_quant_kernel<<<M_DIM, 256, 0, stream>>>(x, nw, Xq, Xs);
    w_abs_partial_kernel<<<4096, 256, 0, stream>>>(w, partials);
    w_scale_final_kernel<<<1, 256, 0, stream>>>(partials, wscale);
    w_quant_kernel<<<(N_DIM * (size_t)K_DIM) / 1024, 256, 0, stream>>>(w, wscale, Wq);

    dim3 grid(N_DIM / 128, M_DIM / 128);
    gemm_i8_wmma_kernel<<<grid, 256, 0, stream>>>(Xq, Wq, Xs, wscale, out);
}
